// MultiHeadAttention_3272765079689
// MI455X (gfx1250) — matrix operand order repaired and checked by reading, not compiled, not run on hardware
//
#include <hip/hip_runtime.h>

// ---------------------------------------------------------------------------
// MI455X / gfx1250, wave32.
//   - All matmuls via v_wmma_f32_16x16x32_bf16 (fp32 accumulate).
//   - Fragment loads = ds_load_b128 pairs (contiguous LDS layouts, 72-elem
//     padded stride for bank spread).
//   - Non-transposed tile staging via Tensor Data Mover (tensor_load_to_lds,
//     TENSORcnt) with hardware padding producing the 72-elem stride;
//     transposed tiles staged element-wise (TDM has no transpose).
// ---------------------------------------------------------------------------

typedef __attribute__((ext_vector_type(16))) __bf16 v16bf;
typedef __attribute__((ext_vector_type(8)))  __bf16 v8bf;
typedef __attribute__((ext_vector_type(8)))  float  v8f;

#define BATCH 4
#define SEQ   1024
#define EMB   1024
#define HEADS 16
#define HDIM  64
#define MTOT  (BATCH * SEQ)   // 4096
#define LSTR  72              // padded LDS row stride (bf16 elems; 144B, 16B-aligned)

#ifndef __has_builtin
#define __has_builtin(x) 0
#endif
#if defined(__AMDGCN__) && __has_builtin(__builtin_amdgcn_tensor_load_to_lds)
#define USE_TDM 1
#else
#define USE_TDM 0
#endif

#if USE_TDM
typedef __attribute__((ext_vector_type(4))) unsigned int v4u_t;
typedef __attribute__((ext_vector_type(4))) int          v4i_t;
typedef __attribute__((ext_vector_type(8))) int          v8i_t;

// DMA a 64x64 bf16 tile (row length 64, arbitrary row stride) global -> LDS,
// with TDM padding of 4 dwords after every 32 dwords => LDS row stride 144B
// (= LSTR elements), matching the fragment-loader layout.
__device__ inline void tdm_load_tile64(const void* gsrc, void* ldst,
                                       unsigned tensor_w, unsigned tensor_h,
                                       unsigned row_stride_elems) {
  const unsigned long long ga = (unsigned long long)(size_t)gsrc;
  const unsigned la = (unsigned)(size_t)ldst;  // generic LDS addr[31:0] = LDS byte offset
  v4u_t g0;
  g0[0] = 1u;                                          // D# count=1, gather off
  g0[1] = la;                                          // lds_addr (bytes)
  g0[2] = (unsigned)ga;                                // global_addr[31:0]
  g0[3] = ((unsigned)(ga >> 32) & 0x01FFFFFFu) | (2u << 30);  // addr[56:32] | type=2
  v8i_t g1;
  g1[0] = (int)0x07110000u;  // data_size=2B, pad_enable, pad_interval=32dw, pad=4dw
  g1[1] = (int)((tensor_w & 0xFFFFu) << 16);           // tensor_dim0[15:0]
  g1[2] = (int)((tensor_w >> 16) | ((tensor_h & 0xFFFFu) << 16));
  g1[3] = (int)((tensor_h >> 16) | (64u << 16));       // tile_dim0 = 64
  g1[4] = 64;                                          // tile_dim1 = 64 (tile_dim2=0)
  g1[5] = (int)row_stride_elems;                       // tensor_dim0_stride[31:0]
  g1[6] = 0;
  g1[7] = 0;
  v4i_t z4 = {0, 0, 0, 0};
#if __clang_major__ >= 23
  v8i_t z8 = {0, 0, 0, 0, 0, 0, 0, 0};
  __builtin_amdgcn_tensor_load_to_lds(g0, g1, z4, z4, z8, 0);
#else
  __builtin_amdgcn_tensor_load_to_lds(g0, g1, z4, z4, 0);
#endif
}
#endif  // USE_TDM

// ---- fragment loaders (CDNA5 16-bit WMMA layouts, wave32) ------------------
// A 16x32 (MxK), tile row-major K-contiguous:
//   lane r(=M), half: K = half*8 + {0..7} and half*8 + {16..23}  -> 2x 16B loads
__device__ inline v16bf load_a_frag(const __bf16* smem, int row0, int lda, int k0) {
  const int lane = threadIdx.x & 31;
  const int r = lane & 15, half = lane >> 4;
  const __bf16* base = smem + (row0 + r) * lda + k0 + half * 8;
  v8bf lo = *(const v8bf*)(base);
  v8bf hi = *(const v8bf*)(base + 16);
  return __builtin_shufflevector(lo, hi, 0, 1, 2, 3, 4, 5, 6, 7,
                                 8, 9, 10, 11, 12, 13, 14, 15);
}

// B 32x16 (KxN), tile staged N-major / K-contiguous (Bt[n][k]):
//   lane n(=N), half: K = half*8 + {0..7} and half*8 + {16..23}  -> 2x 16B loads
__device__ inline v16bf load_b_fragT(const __bf16* smem, int col0, int ldk, int k0) {
  const int lane = threadIdx.x & 31;
  const int n = lane & 15, half = lane >> 4;
  const __bf16* base = smem + (col0 + n) * ldk + k0 + half * 8;
  v8bf lo = *(const v8bf*)(base);
  v8bf hi = *(const v8bf*)(base + 16);
  return __builtin_shufflevector(lo, hi, 0, 1, 2, 3, 4, 5, 6, 7,
                                 8, 9, 10, 11, 12, 13, 14, 15);
}

// ---- fp32 -> bf16 conversion ----------------------------------------------
__global__ __launch_bounds__(256) void f32_to_bf16_kernel(
    const float* __restrict__ in, __bf16* __restrict__ out, int n) {
  int i = blockIdx.x * 256 + threadIdx.x;
  if (i < n) out[i] = (__bf16)in[i];
}

// ---- GEMM: C[M,N] = A[M,K](bf16) @ B[K,N](bf16) + bias ---------------------
// mode 0: write fp32 row-major.   mode 1: write bf16 scattered to [B,H,S,D].
__global__ __launch_bounds__(128) void gemm_bf16_kernel(
    const __bf16* __restrict__ A, const __bf16* __restrict__ B,
    const float* __restrict__ bias, float* __restrict__ Cf,
    __bf16* __restrict__ Cb, int M, int N, int K, int mode) {
  __shared__ __align__(16) __bf16 sA[64 * LSTR];   // [m][k]  K-contiguous
  __shared__ __align__(16) __bf16 sBt[64 * LSTR];  // [n][k]  K-contiguous
  const int m0 = blockIdx.y * 64;
  const int n0 = blockIdx.x * 64;
  const int wave = threadIdx.x >> 5;
  const int lane = threadIdx.x & 31;
  const int r = lane & 15, half = lane >> 4;

  v8f acc[4];
#pragma unroll
  for (int ni = 0; ni < 4; ++ni)
#pragma unroll
    for (int v = 0; v < 8; ++v) acc[ni][v] = 0.0f;

  for (int k0 = 0; k0 < K; k0 += 64) {
#if USE_TDM
    if (threadIdx.x < 32)  // wave 0: DMA A tile via Tensor Data Mover
      tdm_load_tile64(&A[(size_t)m0 * K + k0], sA, (unsigned)K, 64u, (unsigned)K);
#else
    for (int i = threadIdx.x; i < 64 * 64 / 2; i += 128) {
      const int e = i * 2, rr = e >> 6, cc = e & 63;
      *(uint32_t*)&sA[rr * LSTR + cc] =
          *(const uint32_t*)&A[(size_t)(m0 + rr) * K + k0 + cc];
    }
#endif
    // stage B 64x64 transposed: sBt[n][k] = B[k0+k][n0+n]
    for (int i = threadIdx.x; i < 64 * 64; i += 128) {
      const int kk = i >> 6, nn = i & 63;
      sBt[nn * LSTR + kk] = B[(size_t)(k0 + kk) * N + n0 + nn];
    }
    if (k0 + 64 < K)  // global_prefetch_b8 next B tile
      __builtin_prefetch(&B[(size_t)(k0 + 64 + (threadIdx.x >> 1)) * N + n0], 0, 3);
#if USE_TDM
    if (threadIdx.x < 32) __builtin_amdgcn_s_wait_tensorcnt(0);
#endif
    __syncthreads();
#pragma unroll
    for (int ks = 0; ks < 64; ks += 32) {
      const v16bf af = load_a_frag(sA, wave * 16, LSTR, ks);
#pragma unroll
      for (int ni = 0; ni < 4; ++ni) {
        const v16bf bf_ = load_b_fragT(sBt, ni * 16, LSTR, ks);
        acc[ni] = __builtin_amdgcn_wmma_f32_16x16x32_bf16(
            false, af, false, bf_, (short)0, acc[ni], false, false);
      }
    }
    __syncthreads();
  }

#pragma unroll
  for (int ni = 0; ni < 4; ++ni)
#pragma unroll
    for (int v = 0; v < 8; ++v) {
      const int row = m0 + wave * 16 + v + half * 8;
      const int col = n0 + ni * 16 + r;
      const float val = acc[ni][v] + bias[col];
      if (mode == 0) {
        Cf[(size_t)row * N + col] = val;
      } else {
        const int b = row >> 10, s = row & 1023, h = col >> 6, d = col & 63;
        Cb[(((size_t)(b * HEADS + h) * SEQ + s) * HDIM) + d] = (__bf16)val;
      }
    }
}

// ---- causal flash attention with relative-position bias --------------------
// grid: (SEQ/64, BATCH*HEADS), block: 128 (4 waves). Wave w owns rows 16w..16w+15.
__global__ __launch_bounds__(128) void attn_kernel(
    const __bf16* __restrict__ Qg, const __bf16* __restrict__ Kg,
    const __bf16* __restrict__ Vg, const float* __restrict__ relk,
    const float* __restrict__ relv, __bf16* __restrict__ Oc) {
  __shared__ __align__(16) __bf16 sQ[64 * LSTR];   // [row][d]   A-operand
  __shared__ __align__(16) __bf16 sK[64 * LSTR];   // [key][d]   B-operand of QK^T
  __shared__ __align__(16) __bf16 sVt[64 * LSTR];  // [d][key]   B-operand of P@V
  __shared__ __align__(16) __bf16 sP[64 * LSTR];   // [row][key] A-operand
  __shared__ float sQrel[64 * 5];                  // qrel[row][j] = Q[row]·relk[j]

  const int qt = blockIdx.x * 64;
  const int bh = blockIdx.y;
  const int wave = threadIdx.x >> 5;
  const int lane = threadIdx.x & 31;
  const int r = lane & 15, half = lane >> 4;
  const __bf16* Qb = Qg + (size_t)bh * SEQ * HDIM;
  const __bf16* Kb = Kg + (size_t)bh * SEQ * HDIM;
  const __bf16* Vb = Vg + (size_t)bh * SEQ * HDIM;

#if USE_TDM
  if (threadIdx.x < 32) {  // wave 0: DMA Q tile
    tdm_load_tile64(Qb + (size_t)qt * HDIM, sQ, HDIM, SEQ, HDIM);
    __builtin_amdgcn_s_wait_tensorcnt(0);
  }
#else
  for (int i = threadIdx.x; i < 64 * 64 / 2; i += 128) {
    const int e = i * 2, rr = e >> 6, cc = e & 63;
    *(uint32_t*)&sQ[rr * LSTR + cc] =
        *(const uint32_t*)&Qb[(size_t)(qt + rr) * HDIM + cc];
  }
#endif
  __syncthreads();
  for (int idx = threadIdx.x; idx < 64 * 5; idx += 128) {
    const int row = idx / 5, j = idx % 5;
    float acc = 0.f;
    for (int d = 0; d < HDIM; ++d)
      acc += (float)sQ[row * LSTR + d] * relk[j * HDIM + d];
    sQrel[idx] = acc;
  }
  __syncthreads();

  v8f o[4];
  float mrow[8], lrow[8], s0row[8], p1row[8], p2row[8];
#pragma unroll
  for (int v = 0; v < 8; ++v) {
    mrow[v] = -1e30f; lrow[v] = 0.f; s0row[v] = 0.f; p1row[v] = 0.f; p2row[v] = 0.f;
  }
#pragma unroll
  for (int ni = 0; ni < 4; ++ni)
#pragma unroll
    for (int v = 0; v < 8; ++v) o[ni][v] = 0.f;

  const int nkb = blockIdx.x + 1;   // causal: key blocks up to the diagonal
  for (int kb = 0; kb < nkb; ++kb) {
    const int kbase = kb * 64;
#if USE_TDM
    if (threadIdx.x < 32)  // wave 0: DMA K tile (row-major, K-contiguous)
      tdm_load_tile64(Kb + (size_t)kbase * HDIM, sK, HDIM, SEQ, HDIM);
#else
    for (int i = threadIdx.x; i < 64 * 64 / 2; i += 128) {
      const int e = i * 2, rr = e >> 6, cc = e & 63;
      *(uint32_t*)&sK[rr * LSTR + cc] =
          *(const uint32_t*)&Kb[(size_t)(kbase + rr) * HDIM + cc];
    }
#endif
    // V block transposed: sVt[d][key] (key contiguous per d-column)
    for (int i = threadIdx.x; i < 64 * 64; i += 128) {
      const int kk = i >> 6, d = i & 63;
      sVt[d * LSTR + kk] = Vb[(size_t)(kbase + kk) * HDIM + d];
    }
    if (kb + 1 < nkb)  // global_prefetch_b8 next value tile
      __builtin_prefetch(&Vb[(size_t)(kbase + 64 + (threadIdx.x >> 1)) * HDIM], 0, 3);
#if USE_TDM
    if (threadIdx.x < 32) __builtin_amdgcn_s_wait_tensorcnt(0);
#endif
    __syncthreads();

    // scores = Q @ K^T  (16x64 strip per wave)
    v8f sc[4];
#pragma unroll
    for (int ni = 0; ni < 4; ++ni)
#pragma unroll
      for (int v = 0; v < 8; ++v) sc[ni][v] = 0.f;
#pragma unroll
    for (int ks = 0; ks < 64; ks += 32) {
      const v16bf aq = load_a_frag(sQ, wave * 16, LSTR, ks);
#pragma unroll
      for (int ni = 0; ni < 4; ++ni) {
        const v16bf bk = load_b_fragT(sK, ni * 16, LSTR, ks);
        sc[ni] = __builtin_amdgcn_wmma_f32_16x16x32_bf16(
            false, aq, false, bk, (short)0, sc[ni], false, false);
      }
    }

    // rel-k bias + scale + causal mask + online softmax
    float mnew[8], alpha[8], lsum[8], s0sum[8], p1sum[8], p2sum[8];
#pragma unroll
    for (int v = 0; v < 8; ++v) {
      const int row = wave * 16 + v + half * 8;
      const int q = qt + row;
      float mx = -1e30f;
#pragma unroll
      for (int ni = 0; ni < 4; ++ni) {
        const int k = kbase + ni * 16 + r;
        float s;
        if (k > q) {
          s = -1e30f;
        } else {
          int rel = k - q;            // <= 0 here
          rel = rel < -2 ? -2 : rel;  // clip
          s = (sc[ni][v] + sQrel[row * 5 + rel + 2]) * 0.125f;  // /sqrt(64)
        }
        sc[ni][v] = s;
        mx = fmaxf(mx, s);
      }
#pragma unroll
      for (int m = 1; m < 16; m <<= 1) mx = fmaxf(mx, __shfl_xor(mx, m, 32));
      mnew[v] = fmaxf(mrow[v], mx);
      alpha[v] = __expf(mrow[v] - mnew[v]);
      float ls = 0.f, s0 = 0.f, p1 = 0.f, p2 = 0.f;
#pragma unroll
      for (int ni = 0; ni < 4; ++ni) {
        const int k = kbase + ni * 16 + r;
        const float p = (k > q) ? 0.f : __expf(sc[ni][v] - mnew[v]);
        sP[row * LSTR + ni * 16 + r] = (__bf16)p;
        ls += p;
        if (k <= q - 2)      s0 += p;   // rel region 0
        else if (k == q - 1) p1 += p;   // rel region 1
        else if (k == q)     p2 += p;   // rel region 2
      }
#pragma unroll
      for (int m = 1; m < 16; m <<= 1) {
        ls += __shfl_xor(ls, m, 32);
        s0 += __shfl_xor(s0, m, 32);
        p1 += __shfl_xor(p1, m, 32);
        p2 += __shfl_xor(p2, m, 32);
      }
      lsum[v] = ls; s0sum[v] = s0; p1sum[v] = p1; p2sum[v] = p2;
    }

#pragma unroll
    for (int v = 0; v < 8; ++v) {
#pragma unroll
      for (int ni = 0; ni < 4; ++ni) o[ni][v] *= alpha[v];
      lrow[v]  = lrow[v]  * alpha[v] + lsum[v];
      s0row[v] = s0row[v] * alpha[v] + s0sum[v];
      p1row[v] = p1row[v] * alpha[v] + p1sum[v];
      p2row[v] = p2row[v] * alpha[v] + p2sum[v];
      mrow[v]  = mnew[v];
    }

    // O += P @ V (wave reads only its own sP rows: in-wave LDS ordering)
#pragma unroll
    for (int ks = 0; ks < 64; ks += 32) {
      const v16bf ap = load_a_frag(sP, wave * 16, LSTR, ks);
#pragma unroll
      for (int ni = 0; ni < 4; ++ni) {
        const v16bf bv = load_b_fragT(sVt, ni * 16, LSTR, ks);
        o[ni] = __builtin_amdgcn_wmma_f32_16x16x32_bf16(
            false, ap, false, bv, (short)0, o[ni], false, false);
      }
    }
    __syncthreads();
  }

  // finalize: add rel-v contribution, normalize, write concat [B,S,E] bf16
  const int b = bh >> 4, h = bh & 15;
#pragma unroll
  for (int ni = 0; ni < 4; ++ni)
#pragma unroll
    for (int v = 0; v < 8; ++v) {
      const int row = wave * 16 + v + half * 8;
      const int q = qt + row;
      const int d = ni * 16 + r;
      const float val = (o[ni][v]
                         + s0row[v] * relv[0 * HDIM + d]
                         + p1row[v] * relv[1 * HDIM + d]
                         + p2row[v] * relv[2 * HDIM + d]) / lrow[v];
      Oc[((size_t)(b * SEQ + q)) * EMB + h * HDIM + d] = (__bf16)val;
    }
}

// ---------------------------------------------------------------------------
extern "C" void kernel_launch(void* const* d_in, const int* in_sizes, int n_in,
                              void* d_out, int out_size, void* d_ws, size_t ws_size,
                              hipStream_t stream) {
  (void)in_sizes; (void)n_in; (void)out_size; (void)ws_size;
  const float* x    = (const float*)d_in[0];
  const float* W_q  = (const float*)d_in[1];
  const float* b_q  = (const float*)d_in[2];
  const float* W_k  = (const float*)d_in[3];
  const float* b_k  = (const float*)d_in[4];
  const float* W_v  = (const float*)d_in[5];
  const float* b_v  = (const float*)d_in[6];
  const float* W_o  = (const float*)d_in[7];
  const float* b_o  = (const float*)d_in[8];
  const float* relk = (const float*)d_in[9];
  const float* relv = (const float*)d_in[10];

  char* ws = (char*)d_ws;
  __bf16* xb  = (__bf16*)(ws + 0);          // 4096*1024 bf16 = 8 MB
  __bf16* wqb = (__bf16*)(ws + 8388608);    // 2 MB each
  __bf16* wkb = (__bf16*)(ws + 10485760);
  __bf16* wvb = (__bf16*)(ws + 12582912);
  __bf16* wob = (__bf16*)(ws + 14680064);
  __bf16* Qb  = (__bf16*)(ws + 16777216);   // [B,H,S,D] bf16, 8 MB each
  __bf16* Kb  = (__bf16*)(ws + 25165824);
  __bf16* Vb  = (__bf16*)(ws + 33554432);
  __bf16* Ob  = (__bf16*)(ws + 41943040);   // concat head outputs, 8 MB

  const int nx = MTOT * EMB;   // 4194304
  const int nw = EMB * EMB;    // 1048576
  f32_to_bf16_kernel<<<(nx + 255) / 256, 256, 0, stream>>>(x,   xb,  nx);
  f32_to_bf16_kernel<<<(nw + 255) / 256, 256, 0, stream>>>(W_q, wqb, nw);
  f32_to_bf16_kernel<<<(nw + 255) / 256, 256, 0, stream>>>(W_k, wkb, nw);
  f32_to_bf16_kernel<<<(nw + 255) / 256, 256, 0, stream>>>(W_v, wvb, nw);
  f32_to_bf16_kernel<<<(nw + 255) / 256, 256, 0, stream>>>(W_o, wob, nw);

  dim3 ggrid(EMB / 64, MTOT / 64);  // (16, 64)
  gemm_bf16_kernel<<<ggrid, 128, 0, stream>>>(xb, wqb, b_q, nullptr, Qb,
                                              MTOT, EMB, EMB, 1);
  gemm_bf16_kernel<<<ggrid, 128, 0, stream>>>(xb, wkb, b_k, nullptr, Kb,
                                              MTOT, EMB, EMB, 1);
  gemm_bf16_kernel<<<ggrid, 128, 0, stream>>>(xb, wvb, b_v, nullptr, Vb,
                                              MTOT, EMB, EMB, 1);

  attn_kernel<<<dim3(SEQ / 64, BATCH * HEADS), 128, 0, stream>>>(
      Qb, Kb, Vb, relk, relv, Ob);

  gemm_bf16_kernel<<<ggrid, 128, 0, stream>>>(Ob, wob, b_o, (float*)d_out, nullptr,
                                              MTOT, EMB, EMB, 0);
}
